// EncoderDecoder_32117765439796
// MI455X (gfx1250) — compile-verified, matrix operand order checked by reading, not run on hardware
//
#include <hip/hip_runtime.h>

typedef __attribute__((ext_vector_type(16))) __bf16 v16bf;
typedef __attribute__((ext_vector_type(8)))  __bf16 v8bf;
typedef __attribute__((ext_vector_type(8)))  float  v8f;

#define B_TOT 512
#define F_DIM 64
#define H_DIM 256
#define T_LEN 128
#define G3H   768   // 3*H

__device__ __forceinline__ v16bf frag_cat(v8bf lo, v8bf hi) {
  return __builtin_shufflevector(lo, hi, 0,1,2,3,4,5,6,7,8,9,10,11,12,13,14,15);
}

// A-operand fragment (16xK tile, K-slice of 32) from a row-major bf16 buffer.
// ISA 16-bit A 16x32 layout: lane L<16 : row=L,    K = kbase+{0..7, 16..23}
//                            lane L>=16: row=L-16, K = kbase+{8..15,24..31}
__device__ __forceinline__ v16bf load_frag_a(const __bf16* S, int stride, int kbase, int lane) {
  const __bf16* p = S + (lane & 15) * stride + kbase + ((lane >> 4) << 3);
  v8bf lo = *(const v8bf*)(p);
  v8bf hi = *(const v8bf*)(p + 16);
  return frag_cat(lo, hi);
}

// B-operand fragment (Kx16 slice of W^T == 16 contiguous rows of W),
// W row-major (rows x kdim); lane's column = W row (nrow0 + lane&15).
__device__ __forceinline__ v16bf load_frag_b(const __bf16* W, int kdim, int nrow0, int kbase, int lane) {
  const __bf16* p = W + (size_t)(nrow0 + (lane & 15)) * kdim + kbase + ((lane >> 4) << 3);
  v8bf lo = *(const v8bf*)(p);
  v8bf hi = *(const v8bf*)(p + 16);
  return frag_cat(lo, hi);
}

__device__ __forceinline__ v8f wmma_bf16(v16bf a, v16bf b, v8f c) {
  return __builtin_amdgcn_wmma_f32_16x16x32_bf16(false, a, false, b, (short)0, c, false, false);
}

__device__ __forceinline__ v8f bias_bcast(float v) {
  v8f c;
  #pragma unroll
  for (int i = 0; i < 8; ++i) c[i] = v;
  return c;
}

__device__ __forceinline__ float fsigmoid(float v) { return 1.0f / (1.0f + __expf(-v)); }
__device__ __forceinline__ float ftanh_(float v)   { return 2.0f * fsigmoid(2.0f * v) - 1.0f; }

__global__ void cvt_f32_bf16(const float* __restrict__ src, __bf16* __restrict__ dst, int n) {
  int i = blockIdx.x * blockDim.x + threadIdx.x;
  if (i < n) dst[i] = (__bf16)src[i];
}

__global__ __launch_bounds__(256) void gru_encdec(
    const float* __restrict__ x,
    const float* __restrict__ enc_bih, const float* __restrict__ enc_bhh,
    const float* __restrict__ dec_bih, const float* __restrict__ dec_bhh,
    const float* __restrict__ reg_b,
    const __bf16* __restrict__ wWihE, const __bf16* __restrict__ wWhhE,
    const __bf16* __restrict__ wWihD, const __bf16* __restrict__ wWhhD,
    const __bf16* __restrict__ wReg,
    float* __restrict__ out)
{
  __shared__ __bf16 sH[16 * H_DIM];       // hidden state tile (bf16)     8 KB
  __shared__ __bf16 sX[16 * F_DIM];       // current input tile (bf16)    2 KB
  __shared__ float  sRZ[16 * 512];        // r|z pre-activations         32 KB
  __shared__ float  sIN[16 * H_DIM];      // inn                         16 KB
  __shared__ float  sHN[16 * H_DIM];      // hn                          16 KB

  const int tid  = threadIdx.x;
  const int wave = tid >> 5;
  const int lane = tid & 31;
  const int row0 = blockIdx.x * 16;       // global batch row base

  for (int i = tid; i < 16 * H_DIM; i += 256) sH[i] = (__bf16)0.0f;
  __syncthreads();

  for (int phase = 0; phase < 2; ++phase) {
    const __bf16* Wih = phase ? wWihD : wWihE;
    const __bf16* Whh = phase ? wWhhD : wWhhE;
    const float*  bih = phase ? dec_bih : enc_bih;
    const float*  bhh = phase ? dec_bhh : enc_bhh;

    for (int t = 0; t < T_LEN; ++t) {
      // Stage x_t: encoder reads x[:,t,:]; decoder t==0 reads x[:,T_LEN-1,:],
      // later decoder steps reuse the fed-back output already in sX.
      if (phase == 0 || t == 0) {
        const int ts = phase ? (T_LEN - 1) : t;
        for (int i = tid; i < 16 * F_DIM; i += 256) {
          int r = i >> 6, c = i & 63;
          sX[i] = (__bf16)x[((size_t)(row0 + r) * (2 * T_LEN) + ts) * F_DIM + c];
        }
      }
      __syncthreads();

      // Hoist A operands for this step (shared by all 6 tiles of this wave).
      v16bf ax0 = load_frag_a(sX, F_DIM, 0,  lane);
      v16bf ax1 = load_frag_a(sX, F_DIM, 32, lane);
      v16bf ah[8];
      #pragma unroll
      for (int kk = 0; kk < 8; ++kk) ah[kk] = load_frag_a(sH, H_DIM, kk * 32, lane);

      const int mrow = (lane >> 4) << 3;   // D-matrix row base for this lane
      const int ncol = lane & 15;          // D-matrix column for this lane

      #pragma unroll
      for (int ti = 0; ti < 6; ++ti) {
        const int tile = wave * 6 + ti;        // 0..47 over 3H=768 gate cols
        const int ng   = tile * 16 + ncol;     // global gate row 0..767
        if (tile < 32) {                       // r and z gates: gi+gh fused
          v8f acc = bias_bcast(bih[ng] + bhh[ng]);
          acc = wmma_bf16(ax0, load_frag_b(Wih, F_DIM, tile * 16, 0,  lane), acc);
          acc = wmma_bf16(ax1, load_frag_b(Wih, F_DIM, tile * 16, 32, lane), acc);
          #pragma unroll
          for (int kk = 0; kk < 8; ++kk)
            acc = wmma_bf16(ah[kk], load_frag_b(Whh, H_DIM, tile * 16, kk * 32, lane), acc);
          #pragma unroll
          for (int r = 0; r < 8; ++r)
            sRZ[(mrow + r) * 512 + tile * 16 + ncol] = acc[r];
        } else {                               // n gate: keep inn and hn apart
          v8f ai = bias_bcast(bih[ng]);
          ai = wmma_bf16(ax0, load_frag_b(Wih, F_DIM, tile * 16, 0,  lane), ai);
          ai = wmma_bf16(ax1, load_frag_b(Wih, F_DIM, tile * 16, 32, lane), ai);
          v8f an = bias_bcast(bhh[ng]);
          #pragma unroll
          for (int kk = 0; kk < 8; ++kk)
            an = wmma_bf16(ah[kk], load_frag_b(Whh, H_DIM, tile * 16, kk * 32, lane), an);
          const int cb = (tile - 32) * 16 + ncol;
          #pragma unroll
          for (int r = 0; r < 8; ++r) {
            sIN[(mrow + r) * H_DIM + cb] = ai[r];
            sHN[(mrow + r) * H_DIM + cb] = an[r];
          }
        }
      }
      __syncthreads();

      // Element-wise GRU state update: each thread owns 1 row x 16 cols.
      {
        const int r  = tid >> 4;
        const int c0 = (tid & 15) << 4;
        #pragma unroll
        for (int j = 0; j < 16; ++j) {
          const int c = c0 + j;
          float rr = fsigmoid(sRZ[r * 512 + c]);
          float zz = fsigmoid(sRZ[r * 512 + 256 + c]);
          float nn = ftanh_(sIN[r * H_DIM + c] + rr * sHN[r * H_DIM + c]);
          float hp = (float)sH[r * H_DIM + c];
          sH[r * H_DIM + c] = (__bf16)((1.0f - zz) * nn + zz * hp);
        }
      }
      __syncthreads();

      if (phase == 1) {
        // out_t = h_new @ reg_W^T + reg_b ; write f32 out and feed back bf16.
        if (wave < 4) {
          const int ng = wave * 16 + ncol;     // reg_W row 0..63
          v8f acc = bias_bcast(reg_b[ng]);
          #pragma unroll
          for (int kk = 0; kk < 8; ++kk) {
            v16bf a = load_frag_a(sH, H_DIM, kk * 32, lane);
            acc = wmma_bf16(a, load_frag_b(wReg, H_DIM, wave * 16, kk * 32, lane), acc);
          }
          #pragma unroll
          for (int r = 0; r < 8; ++r) {
            const int rl = mrow + r;
            const int cg = wave * 16 + ncol;
            out[((size_t)(row0 + rl) * T_LEN + t) * F_DIM + cg] = acc[r];
            sX[rl * F_DIM + cg] = (__bf16)acc[r];
          }
        }
        // next iteration's top-of-step barrier orders sX feedback vs. reads
      }
    }
  }
}

extern "C" void kernel_launch(void* const* d_in, const int* in_sizes, int n_in,
                              void* d_out, int out_size, void* d_ws, size_t ws_size,
                              hipStream_t stream) {
  (void)in_sizes; (void)n_in; (void)out_size; (void)ws_size;
  const float* x       = (const float*)d_in[0];
  const float* enc_Wih = (const float*)d_in[1];
  const float* enc_Whh = (const float*)d_in[2];
  const float* enc_bih = (const float*)d_in[3];
  const float* enc_bhh = (const float*)d_in[4];
  const float* dec_Wih = (const float*)d_in[5];
  const float* dec_Whh = (const float*)d_in[6];
  const float* dec_bih = (const float*)d_in[7];
  const float* dec_bhh = (const float*)d_in[8];
  const float* reg_W   = (const float*)d_in[9];
  const float* reg_b   = (const float*)d_in[10];

  __bf16* ws    = (__bf16*)d_ws;
  __bf16* wWihE = ws;
  __bf16* wWhhE = wWihE + G3H * F_DIM;
  __bf16* wWihD = wWhhE + G3H * H_DIM;
  __bf16* wWhhD = wWihD + G3H * F_DIM;
  __bf16* wReg  = wWhhD + G3H * H_DIM;

  {
    int n;
    n = G3H * F_DIM;   cvt_f32_bf16<<<(n + 255) / 256, 256, 0, stream>>>(enc_Wih, wWihE, n);
    n = G3H * H_DIM;   cvt_f32_bf16<<<(n + 255) / 256, 256, 0, stream>>>(enc_Whh, wWhhE, n);
    n = G3H * F_DIM;   cvt_f32_bf16<<<(n + 255) / 256, 256, 0, stream>>>(dec_Wih, wWihD, n);
    n = G3H * H_DIM;   cvt_f32_bf16<<<(n + 255) / 256, 256, 0, stream>>>(dec_Whh, wWhhD, n);
    n = F_DIM * H_DIM; cvt_f32_bf16<<<(n + 255) / 256, 256, 0, stream>>>(reg_W, wReg, n);
  }

  gru_encdec<<<B_TOT / 16, 256, 0, stream>>>(
      x, enc_bih, enc_bhh, dec_bih, dec_bhh, reg_b,
      wWihE, wWhhE, wWihD, wWhhD, wReg, (float*)d_out);
}
